// Flow_16269336117643
// MI455X (gfx1250) — matrix operand order repaired and checked by reading, not compiled, not run on hardware
//
#include <hip/hip_runtime.h>

#define HIDDEN 128
#define NLAYERS 8
#define WAVES_PER_BLOCK 8
#define SAMPLES_PER_WAVE 16

typedef __attribute__((ext_vector_type(16))) _Float16 v16h;
typedef __attribute__((ext_vector_type(8)))  float    v8f;

__global__ __launch_bounds__(256) void flow_coupling_kernel(
    const float* __restrict__ z,  const float* __restrict__ W1,
    const float* __restrict__ b1, const float* __restrict__ W2,
    const float* __restrict__ b2, float* __restrict__ out_x,
    float* __restrict__ out_ld, int Bn)
{
    // ---- LDS staging: all weights resident for the whole kernel (~25 KB) ----
    __shared__ __align__(16) float    s_w1a[NLAYERS][HIDDEN];   // active row of W1 per layer
    __shared__ __align__(16) float    s_b1 [NLAYERS][HIDDEN];
    __shared__ __align__(32) _Float16 s_w2t[NLAYERS][4][HIDDEN]; // W2 transposed [l][n][k], f16
    __shared__ __align__(32) _Float16 s_zero[HIDDEN];            // zero row for B cols n>=4
    __shared__ float s_b2[NLAYERS][4];
    __shared__ float s_c[WAVES_PER_BLOCK][256];                  // per-wave C spill [n][m]

    const int tid = threadIdx.x;
    for (int i = tid; i < NLAYERS * HIDDEN; i += 256) {
        int l = i >> 7, k = i & 127;
        s_w1a[l][k] = W1[((l << 1) + (l & 1)) * HIDDEN + k];     // W1[l][l%2][k]
        s_b1[l][k]  = b1[i];
    }
    for (int i = tid; i < NLAYERS * 4 * HIDDEN; i += 256) {
        int l = i >> 9, rem = i & 511, n = rem >> 7, k = rem & 127;
        s_w2t[l][n][k] = (_Float16)W2[(l * HIDDEN + k) * 4 + n]; // W2[l][k][n]
    }
    for (int i = tid; i < HIDDEN; i += 256) s_zero[i] = (_Float16)0.f;
    if (tid < NLAYERS * 4) s_b2[tid >> 2][tid & 3] = b2[tid];
    __syncthreads();

    const int wave = tid >> 5;
    const int lane = tid & 31;
    const int half = lane >> 4;      // which 16-lane half of the wave
    const int m    = lane & 15;      // sample row (A/C), also B-matrix column n
    const int tile = blockIdx.x * WAVES_PER_BLOCK + wave;
    int idx = tile * SAMPLES_PER_WAVE + m;
    if (idx >= Bn) idx = Bn - 1;     // clamp so EXEC stays all-1s for WMMA

    float x0 = z[idx * 2 + 0];
    float x1 = z[idx * 2 + 1];
    float logdet = 0.f;
    float* cs = s_c[wave];

    for (int l = 0; l < NLAYERS; ++l) {
        const int   act   = l & 1;          // active coordinate (kept)
        const int   inact = act ^ 1;        // transformed coordinate
        const float xa    = act ? x1 : x0;
        const float* w1p = s_w1a[l];
        const float* b1p = s_b1[l];
        // Uniform pointer select: columns n<4 read real W2 row, others a zero row.
        const _Float16* w2p = (m < 4) ? &s_w2t[l][m][0] : &s_zero[0];

        v8f c = {};
#pragma unroll
        for (int ch = 0; ch < 4; ++ch) {    // K = 128 in 4 chunks of 32
            const int base = ch * 32;
            const int off0 = base + half * 16;       // K run for v16h elems 0..7
            const int off1 = base + half * 16 + 8;   // K run for v16h elems 8..15
            float wa[16], ba[16];
            *(float4*)&wa[0]  = *(const float4*)(w1p + off0);
            *(float4*)&wa[4]  = *(const float4*)(w1p + off0 + 4);
            *(float4*)&wa[8]  = *(const float4*)(w1p + off1);
            *(float4*)&wa[12] = *(const float4*)(w1p + off1 + 4);
            *(float4*)&ba[0]  = *(const float4*)(b1p + off0);
            *(float4*)&ba[4]  = *(const float4*)(b1p + off0 + 4);
            *(float4*)&ba[8]  = *(const float4*)(b1p + off1);
            *(float4*)&ba[12] = *(const float4*)(b1p + off1 + 4);
            // Rank-1 hid build on VALU (co-executes with the WMMA chain)
            v16h a;
#pragma unroll
            for (int e = 0; e < 16; ++e)
                a[e] = (_Float16)fmaxf(fmaf(xa, wa[e], ba[e]), 0.f);
            // B chunk: 16 consecutive f16 K-values of column n (two ds_load_b128)
            v16h bm = *(const v16h*)(w2p + base + half * 16);
            c = __builtin_amdgcn_wmma_f32_16x16x32_f16(
                    false, a, false, bm, (short)0, c, false, false);
        }
        // Spill C to per-wave LDS scratch in [n][M] order
#pragma unroll
        for (int r = 0; r < 8; ++r)
            cs[m * 16 + (half * 8 + r)] = c[r];
        __builtin_amdgcn_wave_barrier();   // same-wave DS ops are in-order

        const float t  = cs[inact * 16 + m]       + s_b2[l][inact];
        const float sp = cs[(2 + inact) * 16 + m] + s_b2[l][2 + inact];
        const float s  = tanhf(sp);
        const float xi = act ? x0 : x1;           // inactive coordinate value
        const float xn = fmaf(xi, expf(s), t);
        if (act) x0 = xn; else x1 = xn;
        logdet += s;
        __builtin_amdgcn_wave_barrier();
    }

    // Both 16-lane halves hold identical state; only half 0 writes.
    if (half == 0) {
        const int g = tile * SAMPLES_PER_WAVE + m;
        if (g < Bn) {
            out_x[g * 2 + 0] = x0;
            out_x[g * 2 + 1] = x1;
            out_ld[g]        = logdet;
        }
    }
}

extern "C" void kernel_launch(void* const* d_in, const int* in_sizes, int n_in,
                              void* d_out, int out_size, void* d_ws, size_t ws_size,
                              hipStream_t stream) {
    const float* z  = (const float*)d_in[0];
    const float* W1 = (const float*)d_in[1];
    const float* b1 = (const float*)d_in[2];
    const float* W2 = (const float*)d_in[3];
    const float* b2 = (const float*)d_in[4];
    const int Bn = in_sizes[0] / 2;                 // (B, 2) fp32
    float* out_x  = (float*)d_out;                  // x: B*2 floats
    float* out_ld = out_x + (size_t)Bn * 2;         // log_det: B floats
    const int samplesPerBlock = WAVES_PER_BLOCK * SAMPLES_PER_WAVE; // 128
    const int blocks = (Bn + samplesPerBlock - 1) / samplesPerBlock; // 4096
    flow_coupling_kernel<<<blocks, 256, 0, stream>>>(z, W1, b1, W2, b2,
                                                     out_x, out_ld, Bn);
}
